// MultiHeadSelfAttention_10067403342163
// MI455X (gfx1250) — matrix operand order repaired and checked by reading, not compiled, not run on hardware
//
#include <hip/hip_runtime.h>
#include <math.h>

// ---------------------------------------------------------------------------
// MI455X (gfx1250) multi-head self-attention, bf16 WMMA pipeline.
//   B=2, S=2048, D=1024, H=16, hd=64
//   d_out = [ out (B*S*D f32) | attn_weights mean (B*S*S f32) ]
//
// All GEMMs run through v_wmma_f32_16x16x32_bf16 (fp32 accumulate).
// K/V tiles are staged global->LDS with GLOBAL_LOAD_ASYNC_TO_LDS_B128
// (ASYNCcnt) when the builtin is available.
// ---------------------------------------------------------------------------

typedef __attribute__((ext_vector_type(16))) __bf16 v16bf;
typedef __attribute__((ext_vector_type(8)))  __bf16 v8bf;
typedef __attribute__((ext_vector_type(8)))  float  v8f;

#define CB  2
#define CS  2048
#define CD  1024
#define CH  16
#define CHD 64
#define C3D 3072

// ---------------- async global->LDS (CDNA5) --------------------------------

#if defined(__HIP_DEVICE_COMPILE__) && defined(__gfx1250__) && \
    __has_builtin(__builtin_amdgcn_global_load_async_to_lds_b128) && \
    __has_builtin(__builtin_amdgcn_s_wait_asynccnt)
#define HAVE_ASYNC_LDS 1
#else
#define HAVE_ASYNC_LDS 0
#endif

// builtin signature (from hipcc diagnostic): (int4 AS1*, int4 AS3*, imm, imm)
typedef int v4i __attribute__((vector_size(16)));
typedef __attribute__((address_space(1))) v4i g_v4i;
typedef __attribute__((address_space(3))) v4i l_v4i;

// copy 16 bytes global -> LDS (per-lane addresses)
__device__ __forceinline__ void g2l16(const void* g, void* l) {
#if HAVE_ASYNC_LDS
    __builtin_amdgcn_global_load_async_to_lds_b128(
        (g_v4i*)(unsigned long long)g,
        (l_v4i*)(unsigned)(unsigned long long)l, 0, 0);
#else
    *(uint4*)l = *(const uint4*)g;
#endif
}
__device__ __forceinline__ void g2l_join() {
#if HAVE_ASYNC_LDS
    __builtin_amdgcn_s_wait_asynccnt(0);
#endif
}

// ---------------- WMMA helpers ---------------------------------------------

__device__ __forceinline__ v8f wmma_bf16(v16bf a, v16bf b, v8f c) {
    return __builtin_amdgcn_wmma_f32_16x16x32_bf16(
        false, a, false, b, (short)0, c, false, false);
}

// A-operand (16x32 bf16, row-major LDS tile, pitch in halves).
// lanes 0-15: row M=lane, K 0..7 & 16..23; lanes 16-31: row M=lane-16, K 8..15 & 24..31.
__device__ __forceinline__ v16bf load_a_bf(const __bf16* tile, int pitch, int lane) {
    int r = lane & 15, hs = lane >> 4;
    const __bf16* p = tile + r * pitch + hs * 8;
    v8bf lo = *(const v8bf*)(p);
    v8bf hi = *(const v8bf*)(p + 16);
    return __builtin_shufflevector(lo, hi, 0,1,2,3,4,5,6,7,8,9,10,11,12,13,14,15);
}

// A-operand from an fp32 tile (softmax probabilities), converted to bf16.
__device__ __forceinline__ v16bf load_a_f32(const float* tile, int pitch, int ko, int lane) {
    int r = lane & 15, hs = lane >> 4;
    const float* p = tile + (size_t)r * pitch + ko + hs * 8;
    v16bf out;
#pragma unroll
    for (int j = 0; j < 8; ++j) out[j]     = (__bf16)p[j];
#pragma unroll
    for (int j = 0; j < 8; ++j) out[8 + j] = (__bf16)p[16 + j];
    return out;
}

// B-operand from [k][n] LDS tile: lanes 0-15 col N=lane, K 0..7 & 16..23; lanes 16-31 col N=lane-16, K 8..15 & 24..31.
__device__ __forceinline__ v16bf load_b_bf(const __bf16* tile, int pitch, int lane) {
    int c = lane & 15, hs = lane >> 4;
    v16bf out;
#pragma unroll
    for (int j = 0; j < 8; ++j) out[j]     = tile[(hs * 8 + j) * pitch + c];
#pragma unroll
    for (int j = 0; j < 8; ++j) out[8 + j] = tile[(16 + hs * 8 + j) * pitch + c];
    return out;
}

// ---------------- GEMM:  C[M,N] = A[M,K](f32) @ W[N,K](f32)^T + bias -------
// Computed as C^T = W @ A^T so the big W tile stays row-major in LDS
// (vectorized staging) and only the 64x32 x-tile is transposed.
// Block: 256 thr = 8 waves; block tile M=64 x N=128; wave w owns n-tile w
// (16 rows of W) x all four 16-wide m-tiles -> 4 accumulators.

#define WS_PITCH 40   // halves; 16B aligned rows, conflict-free A reads
#define AT_PITCH 72   // halves; 16B aligned lane stride for B reads

template <bool OUT_BF16>
__global__ __launch_bounds__(256) void gemm_nt(const float* __restrict__ A,
                                               const float* __restrict__ W,
                                               const float* __restrict__ bias,
                                               void* __restrict__ Cv,
                                               int M, int N, int K) {
    __shared__ __bf16 Ws[128 * WS_PITCH];   // W tile, row-major [n][k]
    __shared__ __bf16 At[32 * AT_PITCH];    // x tile, transposed [k][m]

    const int t = threadIdx.x, lane = t & 31, w = t >> 5;
    const int rowBase = blockIdx.x * 64;    // m
    const int colBase = blockIdx.y * 128;   // n

    v8f acc[4] = {};

    for (int k0 = 0; k0 < K; k0 += 32) {
        {   // stage W rows (coalesced, f32 -> bf16, 16B stores)
            int n = t >> 1, kc = (t & 1) * 16;
            const float* s = W + (size_t)(colBase + n) * K + k0 + kc;
            v8bf p0, p1;
#pragma unroll
            for (int j = 0; j < 8; ++j) { p0[j] = (__bf16)s[j]; p1[j] = (__bf16)s[8 + j]; }
            *(v8bf*)(Ws + n * WS_PITCH + kc)     = p0;
            *(v8bf*)(Ws + n * WS_PITCH + kc + 8) = p1;
            if (k0 + 32 < K) __builtin_prefetch(s + 32, 0, 0);
        }
        {   // stage x tile transposed: At[k][m]
            int m = t >> 2, kc = (t & 3) * 8;
            const float* s = A + (size_t)(rowBase + m) * K + k0 + kc;
#pragma unroll
            for (int j = 0; j < 8; ++j) At[(kc + j) * AT_PITCH + m] = (__bf16)s[j];
            if (k0 + 32 < K) __builtin_prefetch(s + 32, 0, 0);
        }
        __syncthreads();

        v16bf a = load_a_bf(Ws + (w * 16) * WS_PITCH, WS_PITCH, lane);
#pragma unroll
        for (int mi = 0; mi < 4; ++mi) {
            v16bf b = load_b_bf(At + mi * 16, AT_PITCH, lane);
            acc[mi] = wmma_bf16(a, b, acc[mi]);
        }
        __syncthreads();
    }

    // epilogue: C^T tile -> lane's 8 values are contiguous in n
    const int hs = lane >> 4, ln = lane & 15;
    const int nbase = colBase + w * 16 + 8 * hs;
#pragma unroll
    for (int mi = 0; mi < 4; ++mi) {
        const int m = rowBase + mi * 16 + ln;
        float v[8];
#pragma unroll
        for (int j = 0; j < 8; ++j) v[j] = acc[mi][j] + bias[nbase + j];
        if constexpr (OUT_BF16) {
            v8bf pk;
#pragma unroll
            for (int j = 0; j < 8; ++j) pk[j] = (__bf16)v[j];
            *(v8bf*)((__bf16*)Cv + (size_t)m * N + nbase) = pk;     // 16B store
        } else {
            float* d = (float*)Cv + (size_t)m * N + nbase;
            *(float4*)(d)     = float4{v[0], v[1], v[2], v[3]};
            *(float4*)(d + 4) = float4{v[4], v[5], v[6], v[7]};
        }
    }
}

// ---------------- Attention ------------------------------------------------
// One block per (batch, 16-row q-tile); 256 thr = 8 waves; all 16 heads
// looped inside so the head-mean weights are written once (no atomics).
// scores computed as S^T = K @ Q^T  -> K tile stays row-major (async-staged).
// LDS: sc[16][2052] f32 | KV[256][72] bf16 (K / V / reduce scratch) | QT[64][24] bf16.

#define SC_PITCH 2052
#define SC_BYTES (16 * SC_PITCH * 4)          // 131328
#define KV_PITCH 72
#define KV_BYTES (256 * KV_PITCH * 2)         // 36864
#define QT_PITCH 24
#define QT_BYTES (64 * QT_PITCH * 2)          // 3072
#define ATTN_LDS (SC_BYTES + KV_BYTES + QT_BYTES)

__global__ __launch_bounds__(256) void attention_kernel(
        const unsigned short* __restrict__ qkv_us,   // bf16 [B,S,3D]
        float* __restrict__ ctx,                     // f32  [B,S,D]
        float* __restrict__ outw) {                  // f32  [B,S,S]
    extern __shared__ char smem[];
    float*  sc = (float*)smem;
    __bf16* KV = (__bf16*)(smem + SC_BYTES);
    __bf16* QT = (__bf16*)(smem + SC_BYTES + KV_BYTES);
    const __bf16* qkv = (const __bf16*)qkv_us;

    const int blk = blockIdx.x;
    const int b   = blk >> 7;
    const int q0  = (blk & 127) * 16;
    const int t = threadIdx.x, lane = t & 31, w = t >> 5;
    const int hs = lane >> 4, ln = lane & 15;

    for (int h = 0; h < CH; ++h) {
        // ---- stage Q transposed: QT[d][q] (tiny, once per head) ----
        {
            int q = t & 15, dc = (t >> 4) * 4;
            const __bf16* s = qkv + ((size_t)(b * CS + q0 + q)) * C3D + h * CHD + dc;
#pragma unroll
            for (int j = 0; j < 4; ++j) QT[(dc + j) * QT_PITCH + q] = s[j];
        }
        __syncthreads();

        // ---- scores^T: per 128-key chunk, async-stage K rows then WMMA ----
        for (int kb = 0; kb < 16; ++kb) {
            {   // K rows kb*128 .. +127, row-major -> KV[0..127]
                int kk = t >> 1, c32 = (t & 1) * 32;
                const __bf16* s = qkv + ((size_t)(b * CS + kb * 128 + kk)) * C3D
                                  + CD + h * CHD + c32;
                __bf16* d = KV + kk * KV_PITCH + c32;
#pragma unroll
                for (int j = 0; j < 4; ++j) g2l16(s + j * 8, d + j * 8);
            }
            g2l_join();
            __syncthreads();

            v8f acc = {};
#pragma unroll
            for (int ko = 0; ko < 64; ko += 32) {
                v16bf a  = load_a_bf(KV + (w * 16) * KV_PITCH + ko, KV_PITCH, lane); // A = K rows
                v16bf bq = load_b_bf(QT + ko * QT_PITCH, QT_PITCH, lane);            // B = Q^T
                acc = wmma_bf16(a, bq, acc);
            }
            // C^T: M = key (j + 8*hs), N = q (ln); contiguous in key -> float4 x2
            {
                float* d = sc + (size_t)ln * SC_PITCH + kb * 128 + w * 16 + 8 * hs;
                *(float4*)(d)     = float4{acc[0] * 0.125f, acc[1] * 0.125f,
                                           acc[2] * 0.125f, acc[3] * 0.125f};
                *(float4*)(d + 4) = float4{acc[4] * 0.125f, acc[5] * 0.125f,
                                           acc[6] * 0.125f, acc[7] * 0.125f};
            }
            __syncthreads();
        }

        // ---- exact softmax over 2048; rows w and w+8 per wave ----
#pragma unroll
        for (int rsel = 0; rsel < 2; ++rsel) {
            const int row = w + rsel * 8;
            float* p = sc + (size_t)row * SC_PITCH;
            float m = -1e30f;
#pragma unroll 8
            for (int j = 0; j < 64; ++j) m = fmaxf(m, p[lane + 32 * j]);
            for (int off = 16; off; off >>= 1) m = fmaxf(m, __shfl_xor(m, off, 32));
            float s = 0.f;
#pragma unroll 8
            for (int j = 0; j < 64; ++j) s += __expf(p[lane + 32 * j] - m);
            for (int off = 16; off; off >>= 1) s += __shfl_xor(s, off, 32);
            const float inv = 1.f / s;
            float* wrow = outw + ((size_t)b * CS + q0 + row) * CS;
#pragma unroll 8
            for (int j = 0; j < 64; ++j) {
                int c = lane + 32 * j;
                float pn = __expf(p[c] - m) * inv;
                p[c] = pn;
                float contrib = pn * (1.f / CH);
                if (h == 0) wrow[c] = contrib;
                else        wrow[c] += contrib;
            }
        }
        __syncthreads();

        // ---- ctx = P @ V; 8 waves: d-tile (w&3), K-half (w>>2) ----
        const int d0   = (w & 3) * 16;
        const int ksub = w >> 2;
        v8f acc = {};
        for (int kb2 = 0; kb2 < 8; ++kb2) {          // 256 keys per chunk
            {   // async-stage V rows kb2*256 .. +255, row-major
                const __bf16* s = qkv + ((size_t)(b * CS + kb2 * 256 + t)) * C3D
                                  + 2 * CD + h * CHD;
                __bf16* d = KV + t * KV_PITCH;
#pragma unroll
                for (int j = 0; j < 8; ++j) g2l16(s + j * 8, d + j * 8);
            }
            g2l_join();
            __syncthreads();
#pragma unroll
            for (int kc = 0; kc < 4; ++kc) {
                int koL = ksub * 128 + kc * 32;      // key offset inside chunk
                v16bf a = load_a_f32(sc, SC_PITCH, kb2 * 256 + koL, lane);
                v16bf bv = load_b_bf(KV + koL * KV_PITCH + d0, KV_PITCH, lane);
                acc = wmma_bf16(a, bv, acc);
            }
            __syncthreads();
        }
        // reduce the two K-halves via LDS (reuse KV as f32 scratch)
        float* red = (float*)KV;
        if (w >= 4) {
#pragma unroll
            for (int j = 0; j < 8; ++j)
                red[(size_t)(w - 4) * 256 + (j + 8 * hs) * 16 + ln] = acc[j];
        }
        __syncthreads();
        if (w < 4) {
#pragma unroll
            for (int j = 0; j < 8; ++j) {
                float v = acc[j] + red[(size_t)w * 256 + (j + 8 * hs) * 16 + ln];
                ctx[((size_t)(b * CS + q0 + j + 8 * hs)) * CD + h * CHD + d0 + ln] = v;
            }
        }
        __syncthreads();
    }
}

// ---------------- Residual + LayerNorm -------------------------------------

__global__ __launch_bounds__(256) void resid_ln(const float* __restrict__ x,
                                                const float* __restrict__ attn,
                                                const float* __restrict__ gamma,
                                                const float* __restrict__ beta,
                                                float* __restrict__ out) {
    const int row = blockIdx.x;
    const int t = threadIdx.x, lane = t & 31, w = t >> 5;
    const float* px = x    + (size_t)row * CD;
    const float* pa = attn + (size_t)row * CD;

    float y[4], s = 0.f, sq = 0.f;
#pragma unroll
    for (int j = 0; j < 4; ++j) {
        float v = px[t + 256 * j] + pa[t + 256 * j];
        y[j] = v; s += v; sq += v * v;
    }
    for (int off = 16; off; off >>= 1) { s += __shfl_xor(s, off, 32); sq += __shfl_xor(sq, off, 32); }

    __shared__ float rs[8], rq[8];
    if (lane == 0) { rs[w] = s; rq[w] = sq; }
    __syncthreads();
    if (w == 0) {
        s  = (lane < 8) ? rs[lane] : 0.f;
        sq = (lane < 8) ? rq[lane] : 0.f;
        for (int off = 4; off; off >>= 1) { s += __shfl_xor(s, off, 32); sq += __shfl_xor(sq, off, 32); }
        if (lane == 0) { rs[0] = s; rq[0] = sq; }
    }
    __syncthreads();

    const float mu  = rs[0] * (1.f / CD);
    const float var = rq[0] * (1.f / CD) - mu * mu;
    const float inv = rsqrtf(var + 1e-5f);
#pragma unroll
    for (int j = 0; j < 4; ++j) {
        int c = t + 256 * j;
        out[(size_t)row * CD + c] = (y[j] - mu) * inv * gamma[c] + beta[c];
    }
}

// ---------------- Host launch ----------------------------------------------

extern "C" void kernel_launch(void* const* d_in, const int* in_sizes, int n_in,
                              void* d_out, int out_size, void* d_ws, size_t ws_size,
                              hipStream_t stream) {
    const float* x     = (const float*)d_in[0];
    const float* w_in  = (const float*)d_in[1];
    const float* b_in  = (const float*)d_in[2];
    const float* w_out = (const float*)d_in[3];
    const float* b_out = (const float*)d_in[4];
    const float* gamma = (const float*)d_in[5];
    const float* beta  = (const float*)d_in[6];

    float* out_ln = (float*)d_out;
    float* out_w  = out_ln + (size_t)CB * CS * CD;

    // workspace: qkvb (bf16, 24MB) | ctxf (f32, 16MB); attnf overlaps dead qkvb
    char* ws = (char*)d_ws;
    unsigned short* qkvb = (unsigned short*)ws;
    float* ctxf  = (float*)(ws + (size_t)CB * CS * C3D * 2);
    float* attnf = (float*)ws;

    const int M = CB * CS;  // 4096

    gemm_nt<true><<<dim3(M / 64, C3D / 128), 256, 0, stream>>>(
        x, w_in, b_in, (void*)qkvb, M, C3D, CD);

    attention_kernel<<<dim3(CB * (CS / 16)), 256, ATTN_LDS, stream>>>(
        qkvb, ctxf, out_w);

    gemm_nt<false><<<dim3(M / 64, CD / 128), 256, 0, stream>>>(
        ctxf, w_out, b_out, (void*)attnf, M, CD, CD);

    resid_ln<<<dim3(M), 256, 0, stream>>>(x, attnf, gamma, beta, out_ln);
}
